// RNNModel_57080115363931
// MI455X (gfx1250) — matrix operand order repaired and checked by reading, not compiled, not run on hardware
//
#include <hip/hip_runtime.h>
#include <math.h>
#include <stdint.h>

// ---------- types ----------
typedef __attribute__((ext_vector_type(16))) __bf16 v16bf;
typedef __attribute__((ext_vector_type(8)))  __bf16 v8bf;
typedef __attribute__((ext_vector_type(8)))  float  v8f;

#define S_LEN 2048
#define I_DIM 1024
#define H_DIM 1024
#define O_DIM 32000

__device__ __forceinline__ unsigned short f2bf(float f) {
    unsigned int u = __float_as_uint(f);
    unsigned int lsb = (u >> 16) & 1u;
    u += 0x7fffu + lsb;               // round-to-nearest-even
    return (unsigned short)(u >> 16);
}

// Async global->LDS copy of 16 bytes (CDNA5 GLOBAL_LOAD_ASYNC_TO_LDS_B128,
// tracked by ASYNCcnt). ldsDst: byte offset into LDS (low 32 bits of generic
// shared pointer); gSrc: 64-bit global address.
__device__ __forceinline__ void async_copy_b128(unsigned ldsDst, const void* gSrc) {
    asm volatile("global_load_async_to_lds_b128 %0, %1, off"
                 :: "v"(ldsDst), "v"(gSrc) : "memory");
}
__device__ __forceinline__ void wait_asynccnt0() {
    asm volatile("s_wait_asynccnt 0x0" ::: "memory");
}
__device__ __forceinline__ unsigned lds_off(const void* p) {
    return (unsigned)(uintptr_t)p;    // low 32 bits of generic LDS addr = LDS byte offset
}

// ---------- fp32 -> bf16 conversion (vectorized x4) ----------
__global__ void cvt_f32_bf16(const float* __restrict__ in,
                             unsigned short* __restrict__ out, long n4) {
    long i = (long)blockIdx.x * blockDim.x + threadIdx.x;
    long stride = (long)gridDim.x * blockDim.x;
    for (; i < n4; i += stride) {
        float4 v = ((const float4*)in)[i];
        uint2 p;
        p.x = (unsigned)f2bf(v.x) | ((unsigned)f2bf(v.y) << 16);
        p.y = (unsigned)f2bf(v.z) | ((unsigned)f2bf(v.w) << 16);
        ((uint2*)out)[i] = p;
    }
}

// ---------- state init (h double-buffer + grid barrier) ----------
__global__ void init_state(float* hbuf, unsigned int* bar) {
    int t = threadIdx.x;
    for (int i = t; i < 2 * H_DIM; i += blockDim.x) hbuf[i] = 0.0f;
    if (t < 2) bar[t] = 0u;
}

// ---------- WMMA bf16 GEMM:  C[M,N] = A[M,K] * B[N,K]^T (+bias1+bias2) ----------
// Tile: 128(M) x 128(N) per workgroup, K staged in chunks of 32, double-buffered
// LDS filled by ASYNC global->LDS loads overlapped with the WMMA stream.
// 8 waves: wave (wM=w>>1 in 0..3, wN=w&1 in 0..1) computes 32x64 = 2x4 WMMA frags.
__global__ __launch_bounds__(256)
void gemm_bf16_wmma(const unsigned short* __restrict__ A,   // M x K, bf16 raw
                    const unsigned short* __restrict__ B,   // N x K, bf16 raw
                    float* __restrict__ C,                  // M x N
                    const float* __restrict__ bias1,
                    const float* __restrict__ bias2,
                    int M, int N, int K) {
    __shared__ unsigned short As[2][128][40];  // 32 data + pad (80B row, 16B aligned)
    __shared__ unsigned short Bs[2][128][40];  // B tile stored [N][K]

    const int tid  = threadIdx.x;
    const int lane = tid & 31;
    const int l15  = lane & 15;
    const int hi16 = lane >> 4;               // 0 or 1
    const int wave = tid >> 5;
    const int Mb   = (wave >> 1) * 32;        // wave M base in tile
    const int Nb   = (wave & 1) * 64;         // wave N base in tile

    const int ldRow = tid >> 1;               // 0..127
    const int ldSeg = (tid & 1) * 8;          // 0 or 8 (bf16 elems)

    const unsigned short* aPtr = A + (size_t)(blockIdx.y * 128 + ldRow) * (size_t)K + ldSeg;
    const unsigned short* bPtr = B + (size_t)(blockIdx.x * 128 + ldRow) * (size_t)K + ldSeg;
    const unsigned ldsA0 = lds_off(&As[0][ldRow][ldSeg]);
    const unsigned ldsB0 = lds_off(&Bs[0][ldRow][ldSeg]);
    const unsigned bufStride = (unsigned)(128 * 40 * sizeof(unsigned short));

    v8f acc[2][4];
#pragma unroll
    for (int mi = 0; mi < 2; ++mi)
#pragma unroll
        for (int ni = 0; ni < 4; ++ni)
#pragma unroll
            for (int r = 0; r < 8; ++r) acc[mi][ni][r] = 0.0f;

    const int NK = K >> 5;

    // prologue: async-stage chunk 0 into buffer 0
    async_copy_b128(ldsA0, aPtr);
    async_copy_b128(ldsA0 + 32, aPtr + 16);
    async_copy_b128(ldsB0, bPtr);
    async_copy_b128(ldsB0 + 32, bPtr + 16);
    wait_asynccnt0();
    __syncthreads();

    const int koffA = hi16 * 8;   // A elems 0..7 -> K=koffA.., elems 8..15 -> K=koffA+16..
    const int koffB = hi16 * 8;   // B elems 0..7 -> K=koffB.., elems 8..15 -> K=koffB+16..

    for (int kc = 0; kc < NK; ++kc) {
        const int cur = kc & 1;
        const bool more = (kc + 1) < NK;
        if (more) {
            // async prefetch of chunk kc+1 into the other buffer; runs
            // concurrently with the WMMA stream below.
            const unsigned nb = (unsigned)(cur ^ 1) * bufStride;
            const int kof = (kc + 1) * 32;
            async_copy_b128(ldsA0 + nb, aPtr + kof);
            async_copy_b128(ldsA0 + nb + 32, aPtr + kof + 16);
            async_copy_b128(ldsB0 + nb, bPtr + kof);
            async_copy_b128(ldsB0 + nb + 32, bPtr + kof + 16);
        }

        v16bf af[2];
#pragma unroll
        for (int mi = 0; mi < 2; ++mi) {
            const int row = Mb + mi * 16 + l15;
            v8bf lo = *(const v8bf*)&As[cur][row][koffA];
            v8bf hi = *(const v8bf*)&As[cur][row][koffA + 16];
            af[mi] = __builtin_shufflevector(lo, hi, 0, 1, 2, 3, 4, 5, 6, 7,
                                             8, 9, 10, 11, 12, 13, 14, 15);
        }
        v16bf bfr[4];
#pragma unroll
        for (int ni = 0; ni < 4; ++ni) {
            const int col = Nb + ni * 16 + l15;
            v8bf lo = *(const v8bf*)&Bs[cur][col][koffB];
            v8bf hi = *(const v8bf*)&Bs[cur][col][koffB + 16];
            bfr[ni] = __builtin_shufflevector(lo, hi, 0, 1, 2, 3, 4, 5, 6, 7,
                                              8, 9, 10, 11, 12, 13, 14, 15);
        }
#pragma unroll
        for (int mi = 0; mi < 2; ++mi)
#pragma unroll
            for (int ni = 0; ni < 4; ++ni)
                acc[mi][ni] = __builtin_amdgcn_wmma_f32_16x16x32_bf16(
                    false, af[mi], false, bfr[ni], (short)0, acc[mi][ni],
                    false, false);

        if (more) wait_asynccnt0();
        __syncthreads();
    }

    // epilogue: D layout -> M = r + 8*hi16, N = l15 per frag
#pragma unroll
    for (int mi = 0; mi < 2; ++mi) {
#pragma unroll
        for (int ni = 0; ni < 4; ++ni) {
            const int col = blockIdx.x * 128 + Nb + ni * 16 + l15;
            float badd = 0.0f;
            if (bias1) badd += bias1[col];
            if (bias2) badd += bias2[col];
#pragma unroll
            for (int r = 0; r < 8; ++r) {
                const int row = blockIdx.y * 128 + Mb + mi * 16 + r + hi16 * 8;
                float v = acc[mi][ni][r] + badd;
                __builtin_nontemporal_store(v, &C[(size_t)row * N + col]);
            }
        }
    }
}

// ---------- sequential RNN scan ----------
// 32 persistent workgroups x 256 threads. WG b owns rows [32b, 32b+32).
// W_hh slice kept resident in LDS (32*1024 f32 = 128KB, uses CDNA5's 320KB LDS).
// h double-buffered in global; one L2-atomic grid barrier per step.
__global__ __launch_bounds__(256)
void rnn_scan(const float* __restrict__ Whh,   // H x H f32
              const float* __restrict__ xp,    // S x H f32 (biases included)
              float* __restrict__ hbuf,        // 2 x H f32
              unsigned short* __restrict__ hs, // S x H bf16 out
              unsigned int* __restrict__ bar,
              int steps) {
    extern __shared__ float smem[];
    float* Ws  = smem;                 // 32*1024
    float* hS  = smem + 32 * 1024;     // 1024
    float* red = hS + 1024;            // 256

    const int tid  = threadIdx.x;
    const int base = blockIdx.x * 32;
    const unsigned nWG = gridDim.x;

    // preload W_hh slice (8192 float4)
    {
        const float4* src = (const float4*)(Whh + (size_t)base * H_DIM);
        float4* dst = (float4*)Ws;
        for (int i = tid; i < 8192; i += 256) dst[i] = src[i];
    }
    __syncthreads();

    const int r = tid >> 3;   // row 0..31
    const int q = tid & 7;    // k-slice 0..7 (128 k's each)

    for (int t = 0; t < steps; ++t) {
        const float* hin = hbuf + (t & 1) * H_DIM;
        float* hout      = hbuf + ((t + 1) & 1) * H_DIM;

        ((float4*)hS)[tid] = ((const float4*)hin)[tid];  // 256*4 = 1024 floats
        __syncthreads();

        const float4* wv = (const float4*)(Ws + r * H_DIM + q * 128);
        const float4* hv = ((const float4*)hS) + q * 32;
        float acc = 0.0f;
#pragma unroll 8
        for (int i = 0; i < 32; ++i) {
            float4 w = wv[i], h = hv[i];
            acc = fmaf(w.x, h.x, acc);
            acc = fmaf(w.y, h.y, acc);
            acc = fmaf(w.z, h.z, acc);
            acc = fmaf(w.w, h.w, acc);
        }
        red[tid] = acc;
        __syncthreads();

        if (tid < 32) {
            float s = 0.0f;
#pragma unroll
            for (int j = 0; j < 8; ++j) s += red[tid * 8 + j];
            s += xp[(size_t)t * H_DIM + base + tid];
            float hn = tanhf(s);
            hout[base + tid] = hn;
            hs[(size_t)t * H_DIM + base + tid] = f2bf(hn);
        }
        __syncthreads();

        if (tid == 0) {
            unsigned old = __hip_atomic_fetch_add(&bar[0], 1u, __ATOMIC_ACQ_REL,
                                                  __HIP_MEMORY_SCOPE_AGENT);
            if (old == nWG - 1u) {
                __hip_atomic_store(&bar[0], 0u, __ATOMIC_RELAXED,
                                   __HIP_MEMORY_SCOPE_AGENT);
                __hip_atomic_fetch_add(&bar[1], 1u, __ATOMIC_RELEASE,
                                       __HIP_MEMORY_SCOPE_AGENT);
            } else {
                while (__hip_atomic_load(&bar[1], __ATOMIC_ACQUIRE,
                                         __HIP_MEMORY_SCOPE_AGENT) <
                       (unsigned)(t + 1))
                    __builtin_amdgcn_s_sleep(1);
            }
        }
        __syncthreads();
    }
}

// ---------- host launch ----------
extern "C" void kernel_launch(void* const* d_in, const int* in_sizes, int n_in,
                              void* d_out, int out_size, void* d_ws,
                              size_t ws_size, hipStream_t stream) {
    (void)in_sizes; (void)n_in; (void)out_size; (void)ws_size;
    const float* x   = (const float*)d_in[0];   // [1,S,I]
    const float* Wih = (const float*)d_in[1];   // [H,I]
    const float* Whh = (const float*)d_in[2];   // [H,H]
    const float* bih = (const float*)d_in[3];   // [H]
    const float* bhh = (const float*)d_in[4];   // [H]
    const float* Wfc = (const float*)d_in[5];   // [O,H]
    const float* bfc = (const float*)d_in[6];   // [O]
    float* out = (float*)d_out;                 // [S,O]

    char* ws = (char*)d_ws;
    size_t off = 0;
    auto carve = [&](size_t bytes) -> char* {
        char* p = ws + off;
        off = (off + bytes + 255) & ~(size_t)255;
        return p;
    };
    unsigned short* xbf   = (unsigned short*)carve((size_t)S_LEN * I_DIM * 2);
    unsigned short* Wihbf = (unsigned short*)carve((size_t)H_DIM * I_DIM * 2);
    unsigned short* Wfcbf = (unsigned short*)carve((size_t)O_DIM * H_DIM * 2);
    float*          xpws  = (float*)carve((size_t)S_LEN * H_DIM * 4);
    unsigned short* hsbf  = (unsigned short*)carve((size_t)S_LEN * H_DIM * 2);
    float*          hbuf  = (float*)carve((size_t)2 * H_DIM * 4);
    unsigned int*   bar   = (unsigned int*)carve(64);

    // fp32 -> bf16 conversions
    {
        long n4 = (long)S_LEN * I_DIM / 4;
        cvt_f32_bf16<<<(unsigned)((n4 + 255) / 256), 256, 0, stream>>>(x, xbf, n4);
        n4 = (long)H_DIM * I_DIM / 4;
        cvt_f32_bf16<<<(unsigned)((n4 + 255) / 256), 256, 0, stream>>>(Wih, Wihbf, n4);
        n4 = (long)O_DIM * H_DIM / 4;
        cvt_f32_bf16<<<(unsigned)((n4 + 255) / 256), 256, 0, stream>>>(Wfc, Wfcbf, n4);
    }

    // xp = x @ W_ih^T + b_ih + b_hh   [S,H]
    gemm_bf16_wmma<<<dim3(H_DIM / 128, S_LEN / 128), 256, 0, stream>>>(
        xbf, Wihbf, xpws, bih, bhh, S_LEN, H_DIM, I_DIM);

    // init recurrence state
    init_state<<<1, 512, 0, stream>>>(hbuf, bar);

    // sequential scan: h_{t+1} = tanh(xp_t + h_t @ W_hh^T), emits hs in bf16
    const size_t smemBytes = (size_t)(32 * 1024 + 1024 + 256) * sizeof(float);
    rnn_scan<<<32, 256, smemBytes, stream>>>(Whh, xpws, hbuf, hsbf, bar, S_LEN);

    // out = hs @ W_fc^T + b_fc   [S,O]
    gemm_bf16_wmma<<<dim3(O_DIM / 128, S_LEN / 128), 256, 0, stream>>>(
        hsbf, Wfcbf, out, bfc, nullptr, S_LEN, O_DIM, H_DIM);
}
